// GraphAttention_5377299054622
// MI455X (gfx1250) — matrix operand order repaired and checked by reading, not compiled, not run on hardware
//
#include <hip/hip_runtime.h>
#include <hip/hip_bf16.h>

// ---------------------------------------------------------------------------
// GAT layer on MI455X (gfx1250, wave32, WMMA).
//  - x converted once to bf16 (L2-resident, halves per-head A traffic)
//  - feat = x @ W[h] via v_wmma_f32_16x16x32_bf16, wave tile 32x64
//    (8 WMMA per K-chunk on 2 A + 4 preloaded B fragments),
//    LDS double-buffered, global_prefetch_b8 for the next K chunk.
//  - softmax-by-segment + aggregation via atomics (scatter/gather bound).
// ---------------------------------------------------------------------------

typedef __attribute__((ext_vector_type(16))) __bf16        v16bf;
typedef __attribute__((ext_vector_type(8)))  float         v8f;
typedef __attribute__((ext_vector_type(4)))  float         v4f;
typedef __attribute__((ext_vector_type(4)))  unsigned int  v4u;

#define N_NODES 50000
#define N_EDGES 1600000
#define N_HEADS 8
#define IN_DIM  512
#define OUT_DIM 64
#define OUT_FULL (N_HEADS * OUT_DIM)          // 512
#define LRELU_ALPHA 0.2f

#define TILE_M 256                            // block rows
#define LDA (TILE_M * 40)                     // halves per A buffer
#define LDB (64 * 40)                         // halves per B buffer

static __device__ __forceinline__ unsigned short f2bf(float f) {
    unsigned u = __float_as_uint(f);
    unsigned r = u + 0x7FFFu + ((u >> 16) & 1u);   // round-to-nearest-even
    return (unsigned short)(r >> 16);
}

// order-preserving float <-> uint map for atomicMax on floats
static __device__ __forceinline__ unsigned fenc(float f) {
    unsigned u = __float_as_uint(f);
    return (u & 0x80000000u) ? ~u : (u | 0x80000000u);
}
static __device__ __forceinline__ float fdec(unsigned u) {
    return __uint_as_float((u & 0x80000000u) ? (u & 0x7FFFFFFFu) : ~u);
}

// ---------------------------------------------------------------------------
// 0) zero-init output (25.6M f32), segment max (uint enc), denom
// ---------------------------------------------------------------------------
__global__ void gat_init_kernel(float* __restrict__ out,
                                unsigned* __restrict__ mbits,
                                float* __restrict__ denom) {
    int i = blockIdx.x * blockDim.x + threadIdx.x;
    if (i < N_NODES * OUT_FULL) out[i] = 0.0f;
    if (i < N_HEADS * N_NODES) { mbits[i] = 0u; denom[i] = 0.0f; }
}

// ---------------------------------------------------------------------------
// 1a) x (f32) -> xbf (bf16), 4 elements per thread
// ---------------------------------------------------------------------------
__global__ void gat_xbf_kernel(const float* __restrict__ x,
                               unsigned short* __restrict__ xbf) {
    int t = blockIdx.x * blockDim.x + threadIdx.x;     // N_NODES*IN_DIM/4 threads
    if (t >= (N_NODES * IN_DIM) / 4) return;
    v4f f = *(const v4f*)(x + (size_t)t * 4);
    unsigned short* dp = xbf + (size_t)t * 4;
    dp[0] = f2bf(f[0]); dp[1] = f2bf(f[1]); dp[2] = f2bf(f[2]); dp[3] = f2bf(f[3]);
}

// ---------------------------------------------------------------------------
// 1b) W[h][k][n] (f32) -> Wt[h][n][k] (bf16): column-major B for clean staging
// ---------------------------------------------------------------------------
__global__ void gat_wt_kernel(const float* __restrict__ W,
                              unsigned short* __restrict__ Wt) {
    int t = blockIdx.x * blockDim.x + threadIdx.x;     // 8*64*512 threads
    if (t >= N_HEADS * OUT_DIM * IN_DIM) return;
    int h = t / (OUT_DIM * IN_DIM);
    int r = t - h * (OUT_DIM * IN_DIM);
    int n = r / IN_DIM;
    int k = r - n * IN_DIM;
    Wt[((size_t)(h * OUT_DIM + n)) * IN_DIM + k] =
        f2bf(W[((size_t)h * IN_DIM + k) * OUT_DIM + n]);
}

// ---------------------------------------------------------------------------
// 2) feat[h] = x @ W[h] with WMMA bf16, LDS double-buffered.
//    Block = 256 thr (8 waves). Block tile = 256 rows x 64 cols, one head.
//    Wave tile = 32 x 64: 2 A frags x 4 B frags -> 8 WMMA per 32-wide K chunk.
//    LDS rows padded to 40 halves (80B) against bank conflicts on b128 reads.
// ---------------------------------------------------------------------------
static __device__ __forceinline__ void stage_tile(
        unsigned short* __restrict__ la, unsigned short* __restrict__ lb,
        const unsigned short* __restrict__ xbf, const unsigned short* __restrict__ Wt,
        int row0, int h, int kb, int tid) {
    // A: 256 rows x 32 k = 1024 x 16B chunks, 4 per thread
#pragma unroll
    for (int c0 = 0; c0 < 4; ++c0) {
        int c = tid + c0 * 256;
        int r = c >> 2, q = c & 3;
        int row = row0 + r; if (row >= N_NODES) row = N_NODES - 1;
        *(v4u*)&la[r * 40 + q * 8] =
            *(const v4u*)&xbf[(size_t)row * IN_DIM + kb + q * 8];
    }
    // B: 64 cols x 32 k = 256 x 16B chunks, 1 per thread
    int n = tid >> 2, q = tid & 3;
    *(v4u*)&lb[n * 40 + q * 8] =
        *(const v4u*)&Wt[((size_t)(h * OUT_DIM + n)) * IN_DIM + kb + q * 8];
}

__global__ void __launch_bounds__(256)
gat_gemm_kernel(const unsigned short* __restrict__ xbf,
                const unsigned short* __restrict__ Wt,
                float* __restrict__ feat) {
    __shared__ unsigned short lds_a[2 * LDA];
    __shared__ unsigned short lds_b[2 * LDB];

    const int h    = blockIdx.y;
    const int row0 = blockIdx.x * TILE_M;
    const int tid  = threadIdx.x;
    const int lane = tid & 31;
    const int wave = tid >> 5;
    const int mlan = lane & 15;   // row-in-16-tile / column-in-n-tile
    const int hi   = lane >> 4;   // which half of the K stripe

    v8f acc0[4] = {{}, {}, {}, {}};   // rows wave*32 + 0..15
    v8f acc1[4] = {{}, {}, {}, {}};   // rows wave*32 + 16..31

    // prologue: stage first K chunk into buffer 0
    stage_tile(lds_a, lds_b, xbf, Wt, row0, h, 0, tid);

    for (int kb = 0; kb < IN_DIM; kb += 32) {
        const int buf = (kb >> 5) & 1;
        __syncthreads();   // staged(buf) visible; prev compute done

        if (kb + 32 < IN_DIM)
            stage_tile(lds_a + (buf ^ 1) * LDA, lds_b + (buf ^ 1) * LDB,
                       xbf, Wt, row0, h, kb + 32, tid);

        // prefetch x two chunks ahead (global_prefetch_b8), one row per thread
        if (kb + 64 < IN_DIM) {
            int pr = row0 + tid; if (pr >= N_NODES) pr = N_NODES - 1;
            __builtin_prefetch(&xbf[(size_t)pr * IN_DIM + kb + 64], 0, 1);
        }

        // A fragments: row m; elements 0..7 -> k=hi*8+e, 8..15 -> k=16+hi*8+e
        union Frag { v16bf v; v4u q[2]; };
        Frag a0, a1;
        {
            const unsigned short* pa = &lds_a[buf * LDA + (wave * 32 + mlan) * 40];
            a0.q[0] = *(const v4u*)(pa + hi * 8);
            a0.q[1] = *(const v4u*)(pa + 16 + hi * 8);
            pa += 16 * 40;
            a1.q[0] = *(const v4u*)(pa + hi * 8);
            a1.q[1] = *(const v4u*)(pa + 16 + hi * 8);
        }
        // B fragments (all 4 preloaded): col n; elements 0..7 -> k=hi*8+e, 8..15 -> k=16+hi*8+e
        Frag b[4];
#pragma unroll
        for (int t = 0; t < 4; ++t) {
            const unsigned short* pb = &lds_b[buf * LDB + (t * 16 + mlan) * 40 + hi * 8];
            b[t].q[0] = *(const v4u*)(pb);
            b[t].q[1] = *(const v4u*)(pb + 16);
        }
#pragma unroll
        for (int t = 0; t < 4; ++t)
            acc0[t] = __builtin_amdgcn_wmma_f32_16x16x32_bf16(
                false, a0.v, false, b[t].v, (short)0, acc0[t], false, false);
#pragma unroll
        for (int t = 0; t < 4; ++t)
            acc1[t] = __builtin_amdgcn_wmma_f32_16x16x32_bf16(
                false, a1.v, false, b[t].v, (short)0, acc1[t], false, false);
    }

    // epilogue: VGPR r -> row_local = 8*hi + r (+16 for acc1), col = mlan + 16*t
    const int rbase = row0 + wave * 32 + hi * 8;
    float* outp = feat + ((size_t)h * N_NODES + rbase) * OUT_DIM + mlan;
    if (row0 + TILE_M <= N_NODES) {       // fast path: 195 of 196 blocks
#pragma unroll
        for (int t = 0; t < 4; ++t)
#pragma unroll
            for (int r = 0; r < 8; ++r) {
                outp[(size_t)r * OUT_DIM + t * 16]        = acc0[t][r];
                outp[(size_t)(16 + r) * OUT_DIM + t * 16] = acc1[t][r];
            }
    } else {
#pragma unroll
        for (int t = 0; t < 4; ++t)
#pragma unroll
            for (int r = 0; r < 8; ++r) {
                if (rbase + r < N_NODES)
                    outp[(size_t)r * OUT_DIM + t * 16] = acc0[t][r];
                if (rbase + 16 + r < N_NODES)
                    outp[(size_t)(16 + r) * OUT_DIM + t * 16] = acc1[t][r];
            }
    }
}

// ---------------------------------------------------------------------------
// 3) alpha_s[h][i] = feat[h][i] . a1[h], alpha_d[h][i] = feat[h][i] . a2[h]
// ---------------------------------------------------------------------------
__global__ void gat_alpha_kernel(const float* __restrict__ feat,
                                 const float* __restrict__ att_w,
                                 float* __restrict__ alpha_s,
                                 float* __restrict__ alpha_d) {
    int t = blockIdx.x * blockDim.x + threadIdx.x;
    if (t >= N_HEADS * N_NODES) return;
    int h = t / N_NODES;
    int node = t - h * N_NODES;
    const float* f  = feat + ((size_t)h * N_NODES + node) * OUT_DIM;
    const float* a1 = att_w + (size_t)h * (2 * OUT_DIM);
    const float* a2 = a1 + OUT_DIM;
    float s1 = 0.0f, s2 = 0.0f;
#pragma unroll
    for (int j = 0; j < OUT_DIM; j += 4) {
        v4f fv = *(const v4f*)(f + j);
        s1 += fv[0]*a1[j] + fv[1]*a1[j+1] + fv[2]*a1[j+2] + fv[3]*a1[j+3];
        s2 += fv[0]*a2[j] + fv[1]*a2[j+1] + fv[2]*a2[j+2] + fv[3]*a2[j+3];
    }
    alpha_s[t] = s1;
    alpha_d[t] = s2;
}

// ---------------------------------------------------------------------------
// 4) segment max of leaky-relu logits over src (atomicMax on encoded uint)
// ---------------------------------------------------------------------------
__global__ void gat_edge_max_kernel(const int* __restrict__ src,
                                    const int* __restrict__ dst,
                                    const float* __restrict__ alpha_s,
                                    const float* __restrict__ alpha_d,
                                    unsigned* __restrict__ mbits) {
    int e = blockIdx.x * blockDim.x + threadIdx.x;
    if (e >= N_EDGES) return;
    int s = src[e], d = dst[e];
#pragma unroll
    for (int h = 0; h < N_HEADS; ++h) {
        float l = alpha_s[h * N_NODES + s] + alpha_d[h * N_NODES + d];
        l = (l >= 0.0f) ? l : LRELU_ALPHA * l;
        atomicMax(&mbits[h * N_NODES + s], fenc(l));
    }
}

// ---------------------------------------------------------------------------
// 5) denom[h][s] = sum exp(logit - m[s])
// ---------------------------------------------------------------------------
__global__ void gat_edge_sum_kernel(const int* __restrict__ src,
                                    const int* __restrict__ dst,
                                    const float* __restrict__ alpha_s,
                                    const float* __restrict__ alpha_d,
                                    const unsigned* __restrict__ mbits,
                                    float* __restrict__ denom) {
    int e = blockIdx.x * blockDim.x + threadIdx.x;
    if (e >= N_EDGES) return;
    int s = src[e], d = dst[e];
#pragma unroll
    for (int h = 0; h < N_HEADS; ++h) {
        float l = alpha_s[h * N_NODES + s] + alpha_d[h * N_NODES + d];
        l = (l >= 0.0f) ? l : LRELU_ALPHA * l;
        float m = fdec(mbits[h * N_NODES + s]);
        atomicAdd(&denom[h * N_NODES + s], __expf(l - m));
    }
}

// ---------------------------------------------------------------------------
// 6) out[s][h*64+j] += (e/denom) * feat[h][d][j]  — 512 thr/block = 8 heads x 64
// ---------------------------------------------------------------------------
__global__ void __launch_bounds__(512)
gat_agg_kernel(const int* __restrict__ src,
               const int* __restrict__ dst,
               const float* __restrict__ alpha_s,
               const float* __restrict__ alpha_d,
               const unsigned* __restrict__ mbits,
               const float* __restrict__ denom,
               const float* __restrict__ feat,
               float* __restrict__ out) {
    __shared__ float att_sh[N_HEADS];
    int e = blockIdx.x;
    int tid = threadIdx.x;
    int s = src[e], d = dst[e];
    if (tid < N_HEADS) {
        int h = tid;
        float l = alpha_s[h * N_NODES + s] + alpha_d[h * N_NODES + d];
        l = (l >= 0.0f) ? l : LRELU_ALPHA * l;
        float m = fdec(mbits[h * N_NODES + s]);
        att_sh[h] = __expf(l - m) / denom[h * N_NODES + s];
    }
    __syncthreads();
    int h = tid >> 6;
    int j = tid & 63;
    float v = att_sh[h] * feat[((size_t)h * N_NODES + d) * OUT_DIM + j];
    atomicAdd(&out[(size_t)s * OUT_FULL + h * OUT_DIM + j], v);
}

// ---------------------------------------------------------------------------
// 7) final ReLU in place
// ---------------------------------------------------------------------------
__global__ void gat_relu_kernel(float* __restrict__ out) {
    int i = blockIdx.x * blockDim.x + threadIdx.x;
    if (i < N_NODES * OUT_FULL) out[i] = fmaxf(out[i], 0.0f);
}

// ---------------------------------------------------------------------------
extern "C" void kernel_launch(void* const* d_in, const int* in_sizes, int n_in,
                              void* d_out, int out_size, void* d_ws, size_t ws_size,
                              hipStream_t stream) {
    const float* x     = (const float*)d_in[0];
    const float* W     = (const float*)d_in[1];
    const float* att_w = (const float*)d_in[2];
    const int*   src   = (const int*)d_in[3];
    const int*   dst   = (const int*)d_in[4];
    float* out = (float*)d_out;

    // workspace carve-up (256B aligned)
    auto align_up = [](size_t v) { return (v + 255) & ~(size_t)255; };
    char* ws = (char*)d_ws;
    float*          feat    = (float*)ws;           ws += align_up((size_t)N_HEADS * N_NODES * OUT_DIM * 4);
    float*          alpha_s = (float*)ws;           ws += align_up((size_t)N_HEADS * N_NODES * 4);
    float*          alpha_d = (float*)ws;           ws += align_up((size_t)N_HEADS * N_NODES * 4);
    unsigned*       mbits   = (unsigned*)ws;        ws += align_up((size_t)N_HEADS * N_NODES * 4);
    float*          denom   = (float*)ws;           ws += align_up((size_t)N_HEADS * N_NODES * 4);
    unsigned short* Wt      = (unsigned short*)ws;  ws += align_up((size_t)N_HEADS * OUT_DIM * IN_DIM * 2);
    unsigned short* xbf     = (unsigned short*)ws;  ws += align_up((size_t)N_NODES * IN_DIM * 2);

    const int outElems = N_NODES * OUT_FULL;

    gat_init_kernel<<<(outElems + 255) / 256, 256, 0, stream>>>(out, mbits, denom);

    gat_xbf_kernel<<<((N_NODES * IN_DIM / 4) + 255) / 256, 256, 0, stream>>>(x, xbf);

    gat_wt_kernel<<<(N_HEADS * OUT_DIM * IN_DIM + 255) / 256, 256, 0, stream>>>(W, Wt);

    dim3 ggrid((N_NODES + TILE_M - 1) / TILE_M, N_HEADS);
    gat_gemm_kernel<<<ggrid, 256, 0, stream>>>(xbf, Wt, feat);

    gat_alpha_kernel<<<(N_HEADS * N_NODES + 255) / 256, 256, 0, stream>>>(feat, att_w, alpha_s, alpha_d);

    gat_edge_max_kernel<<<(N_EDGES + 255) / 256, 256, 0, stream>>>(src, dst, alpha_s, alpha_d, mbits);

    gat_edge_sum_kernel<<<(N_EDGES + 255) / 256, 256, 0, stream>>>(src, dst, alpha_s, alpha_d, mbits, denom);

    gat_agg_kernel<<<N_EDGES, 512, 0, stream>>>(src, dst, alpha_s, alpha_d, mbits, denom, feat, out);

    gat_relu_kernel<<<(outElems + 255) / 256, 256, 0, stream>>>(out);
}
